// BinarizedLinear_90099823935564
// MI455X (gfx1250) — 8-bit matrix operand order repaired and checked by reading, compiled, not run on hardware
//
#include <hip/hip_runtime.h>
#include <hip/hip_bf16.h>
#include <stdint.h>

typedef int   v4i  __attribute__((ext_vector_type(4)));
typedef int   v16i __attribute__((ext_vector_type(16)));
typedef float v8f  __attribute__((ext_vector_type(8)));

#define B_DIM   8192
#define IN_DIM  4096
#define OUT_DIM 4096
#define BM 128
#define BN 128
#define BK 128
#define TILE (BM * BK)
#define KITERS (IN_DIM / BK)

#define FP8_P1 0x38u
#define FP8_M1 0xB8u

#if defined(__HIP_DEVICE_COMPILE__) && \
    __has_builtin(__builtin_amdgcn_global_load_async_to_lds_b128) && \
    __has_builtin(__builtin_amdgcn_s_wait_asynccnt)
#define USE_ASYNC 1
typedef __attribute__((address_space(1))) v4i* glb_v4i_p;
typedef __attribute__((address_space(3))) v4i* lds_v4i_p;
#else
#define USE_ASYNC 0
#endif

__global__ void binarize_kernel(const float* __restrict__ src,
                                uint32_t* __restrict__ dst, int n4) {
    int stride = gridDim.x * blockDim.x;
    for (int i = blockIdx.x * blockDim.x + threadIdx.x; i < n4; i += stride) {
        float4 v = ((const float4*)src)[i];
        uint32_t b0 = (v.x > 0.f) ? FP8_P1 : ((v.x < 0.f) ? FP8_M1 : 0u);
        uint32_t b1 = (v.y > 0.f) ? FP8_P1 : ((v.y < 0.f) ? FP8_M1 : 0u);
        uint32_t b2 = (v.z > 0.f) ? FP8_P1 : ((v.z < 0.f) ? FP8_M1 : 0u);
        uint32_t b3 = (v.w > 0.f) ? FP8_P1 : ((v.w < 0.f) ? FP8_M1 : 0u);
        dst[i] = b0 | (b1 << 8) | (b2 << 16) | (b3 << 24);
    }
}

__global__ void zero_kernel(float* __restrict__ p, int n) {
    int i = blockIdx.x * blockDim.x + threadIdx.x;
    if (i < n) p[i] = 0.f;
}

__global__ __launch_bounds__(256) void bgemm_kernel(
    const int8_t* __restrict__ A,
    const int8_t* __restrict__ W,
    float* __restrict__ out,
    float* __restrict__ gsum,
    float* __restrict__ gsumsq) {
    __shared__ __align__(16) int8_t As[2 * TILE];
    __shared__ __align__(16) int8_t Bs[2 * TILE];

    const int t     = threadIdx.x;
    const int lane  = t & 31;
    const int wave  = t >> 5;
    const int waveM = wave >> 1;
    const int waveN = wave & 1;
    const int mBlock = blockIdx.y;
    const int nBlock = blockIdx.x;

    const size_t aBase = (size_t)(mBlock * BM) * IN_DIM;
    const size_t wBase = (size_t)(nBlock * BN) * IN_DIM;

    auto stage = [&](int bufOff, int k0) {
#pragma unroll
        for (int i = 0; i < 4; ++i) {
            int c   = t + i * 256;
            int row = c >> 3;
            int sg  = (c & 7) * 16;
            const int8_t* ga = &A[aBase + (size_t)row * IN_DIM + k0 + sg];
            const int8_t* gb = &W[wBase + (size_t)row * IN_DIM + k0 + sg];
            int8_t* la = &As[bufOff + row * BK + sg];
            int8_t* lb = &Bs[bufOff + row * BK + sg];
#if USE_ASYNC
            __builtin_amdgcn_global_load_async_to_lds_b128(
                (glb_v4i_p)ga, (lds_v4i_p)la, 0, 0);
            __builtin_amdgcn_global_load_async_to_lds_b128(
                (glb_v4i_p)gb, (lds_v4i_p)lb, 0, 0);
#else
            *(uint4*)la = *(const uint4*)ga;
            *(uint4*)lb = *(const uint4*)gb;
#endif
        }
    };

    v8f acc[2][4] = {};

    const int r  = lane & 15;
    const int hs = lane >> 4;

    stage(0, 0);
    int curOff = 0;

#pragma unroll 1
    for (int kk = 0; kk < KITERS; ++kk) {
#if USE_ASYNC
        __builtin_amdgcn_s_wait_asynccnt(0);
#endif
        __syncthreads();
        if (kk + 1 < KITERS) stage(curOff ^ TILE, (kk + 1) * BK);

        const int8_t* Ac = As + curOff;
        const int8_t* Bc = Bs + curOff;

        union AF { v16i v; int64_t l[8]; } a[2];
#pragma unroll
        for (int mt = 0; mt < 2; ++mt) {
            const int8_t* ap = &Ac[(waveM * 32 + mt * 16 + r) * BK + hs * 16];
#pragma unroll
            for (int u = 0; u < 4; ++u) {
                a[mt].l[u]     = *(const int64_t*)(ap + (u >> 1) * 32 + (u & 1) * 8);
                a[mt].l[u + 4] = *(const int64_t*)(ap + 64 + (u >> 1) * 32 + (u & 1) * 8);
            }
        }
        union BF { v16i v; uint4 q[4]; } b[4];
#pragma unroll
        for (int nt = 0; nt < 4; ++nt) {
            const int8_t* bp = &Bc[(waveN * 64 + nt * 16 + r) * BK + hs * 16];
#pragma unroll
            for (int u = 0; u < 4; ++u) b[nt].q[u] = *(const uint4*)(bp + u * 32);
        }
#pragma unroll
        for (int nt = 0; nt < 4; ++nt) {
            acc[0][nt] = __builtin_amdgcn_wmma_f32_16x16x128_fp8_fp8(
                a[0].v, b[nt].v, (short)0, acc[0][nt], false, false);
            acc[1][nt] = __builtin_amdgcn_wmma_f32_16x16x128_fp8_fp8(
                a[1].v, b[nt].v, (short)0, acc[1][nt], false, false);
        }
        curOff ^= TILE;
    }

    float* lsum   = (float*)&As[0];
    float* lsumsq = lsum + BN;
    __syncthreads();
    if (t < BN) { lsum[t] = 0.f; lsumsq[t] = 0.f; }
    __syncthreads();

    const int mRowBase = mBlock * BM + waveM * 32 + hs * 8;
    const int nColBase = nBlock * BN + waveN * 64 + r;
    const int lcBase   = waveN * 64 + r;

#pragma unroll
    for (int mt = 0; mt < 2; ++mt) {
#pragma unroll
        for (int nt = 0; nt < 4; ++nt) {
            int grow = mRowBase + mt * 16;
            int gcol = nColBase + nt * 16;
            int lcol = lcBase + nt * 16;
            float s = 0.f, s2 = 0.f;
#pragma unroll
            for (int j = 0; j < 8; ++j) {
                float v = acc[mt][nt][j];
                out[(size_t)(grow + j) * OUT_DIM + gcol] = v;
                s += v; s2 += v * v;
            }
            atomicAdd(&lsum[lcol], s);
            atomicAdd(&lsumsq[lcol], s2);
        }
    }

    __syncthreads();
    if (t < BN) {
        atomicAdd(&gsum[nBlock * BN + t], lsum[t]);
        atomicAdd(&gsumsq[nBlock * BN + t], lsumsq[t]);
    }
}

__global__ void finalize_kernel(const float* __restrict__ gsum,
                                const float* __restrict__ gsumsq,
                                float* __restrict__ mean,
                                float* __restrict__ rstd) {
    int c = blockIdx.x * blockDim.x + threadIdx.x;
    if (c < OUT_DIM) {
        float m = gsum[c] * (1.f / B_DIM);
        float v = gsumsq[c] * (1.f / B_DIM) - m * m;
        mean[c] = m;
        rstd[c] = rsqrtf(v + 1e-5f);
    }
}

__global__ void normalize_kernel(float* __restrict__ out,
                                 const float* __restrict__ mean,
                                 const float* __restrict__ rstd) {
    size_t i = (size_t)blockIdx.x * blockDim.x + threadIdx.x;
    int c4 = (int)(i & (OUT_DIM / 4 - 1)) * 4;
    float4 v  = ((const float4*)out)[i];
    float4 m  = *(const float4*)&mean[c4];
    float4 rs = *(const float4*)&rstd[c4];
    v.x = (v.x - m.x) * rs.x;
    v.y = (v.y - m.y) * rs.y;
    v.z = (v.z - m.z) * rs.z;
    v.w = (v.w - m.w) * rs.w;
    ((float4*)out)[i] = v;
}

extern "C" void kernel_launch(void* const* d_in, const int* in_sizes, int n_in,
                              void* d_out, int out_size, void* d_ws, size_t ws_size,
                              hipStream_t stream) {
    const float* x = (const float*)d_in[0];
    const float* w = (const float*)d_in[1];
    float* out = (float*)d_out;

    uint8_t* ws = (uint8_t*)d_ws;
    int8_t* Ai = (int8_t*)ws;
    int8_t* Wi = (int8_t*)(ws + (size_t)B_DIM * IN_DIM);
    float* gsum   = (float*)(ws + (size_t)B_DIM * IN_DIM + (size_t)OUT_DIM * IN_DIM);
    float* gsumsq = gsum + OUT_DIM;
    float* mean   = gsumsq + OUT_DIM;
    float* rstd   = mean + OUT_DIM;

    binarize_kernel<<<4096, 256, 0, stream>>>(x, (uint32_t*)Ai, B_DIM * IN_DIM / 4);
    binarize_kernel<<<4096, 256, 0, stream>>>(w, (uint32_t*)Wi, OUT_DIM * IN_DIM / 4);
    zero_kernel<<<(2 * OUT_DIM + 255) / 256, 256, 0, stream>>>(gsum, 2 * OUT_DIM);

    dim3 grid(OUT_DIM / BN, B_DIM / BM);
    bgemm_kernel<<<grid, 256, 0, stream>>>(Ai, Wi, out, gsum, gsumsq);

    finalize_kernel<<<OUT_DIM / 256, 256, 0, stream>>>(gsum, gsumsq, mean, rstd);
    normalize_kernel<<<(size_t)B_DIM * OUT_DIM / 4 / 256, 256, 0, stream>>>(out, mean, rstd);
}
